// DistanceGraphConvolution_6777458393356
// MI455X (gfx1250) — compile-verified, matrix operand order checked by reading, not run on hardware
//
#include <hip/hip_runtime.h>

// ---------------------------------------------------------------------------
// GCN layer on MI455X (gfx1250, wave32, WMMA 16x16x32 f16 with f32 accum)
//   A      = exp(-0.2 * dist)            [B,N,N], B=8, N=2048
//   dinv   = rsqrt(rowsum(A))            [B,N]
//   out    = (diag(dinv) A diag(dinv) @ x) @ W
// Phase 1: degree kernel (pure f32, bandwidth bound).
// Phase 2: fused normalized-A @ x @ W, A recomputed on the fly in registers,
//          x staged in LDS as f16 with the column scale dinv[k] folded in.
// ---------------------------------------------------------------------------

#define NQ 2048
#define FQ 64
#define XS_STRIDE 40   // halves per f-row (32 + pad), 80B => 16B aligned, bank-spread
#define HP_STRIDE 72   // floats per h-row (64 + pad), 288B => 16B aligned
#define WT_STRIDE 72   // halves per o-row (64 + pad), 144B => 16B aligned

typedef __attribute__((ext_vector_type(16))) _Float16 v16h;
typedef __attribute__((ext_vector_type(8)))  _Float16 v8h;
typedef __attribute__((ext_vector_type(8)))  float    v8f;

__device__ __forceinline__ v16h cat8(v8h lo, v8h hi) {
    return __builtin_shufflevector(lo, hi, 0,1,2,3,4,5,6,7,8,9,10,11,12,13,14,15);
}

// ---------------------------------------------------------------------------
// Phase 1: dinv[row] = rsqrt( sum_k exp(-0.2 * dist[row][k]) )
// One wave per row of 2048 f32; fully coalesced float4 loads.
// ---------------------------------------------------------------------------
__global__ __launch_bounds__(256)
void dgc_degree_kernel(const float* __restrict__ dist, float* __restrict__ dinv) {
    const int lane = threadIdx.x & 31;
    const int w    = threadIdx.x >> 5;
    const int row  = blockIdx.x * 8 + w;                 // 0 .. B*N-1
    const float* dr = dist + (size_t)row * NQ;

    float s = 0.0f;
    #pragma unroll
    for (int it = 0; it < 16; ++it) {
        float4 v = *(const float4*)(dr + it * 128 + lane * 4);
        s += __expf(-0.2f * v.x) + __expf(-0.2f * v.y) +
             __expf(-0.2f * v.z) + __expf(-0.2f * v.w);
    }
    #pragma unroll
    for (int off = 16; off > 0; off >>= 1) s += __shfl_xor(s, off, 32);
    if (lane == 0) dinv[row] = rsqrtf(s);
}

// ---------------------------------------------------------------------------
// Phase 2: out = (diag(dinv) A diag(dinv) @ x) @ W
// Grid: 8 batches * 32 row-tiles of 64 rows. Block: 256 threads = 8 waves.
// Waves: 4 M-groups (16 rows each) x 2 K-halves (1024 each).
// ---------------------------------------------------------------------------
__global__ __launch_bounds__(256)
void dgc_gcn_kernel(const float* __restrict__ x,
                    const float* __restrict__ dist,
                    const float* __restrict__ W,
                    const float* __restrict__ dinv_g,
                    float* __restrict__ out) {
    __shared__ __align__(16) float    s_dinv[NQ];                 //  8 KB
    __shared__ __align__(16) _Float16 s_xs[2][2][FQ * XS_STRIDE]; // 20 KB (khalf, dbuf)
    __shared__ __align__(16) float    s_hp[8][16 * HP_STRIDE];    // 36 KB partial h
    __shared__ __align__(16) _Float16 s_wt[FQ * WT_STRIDE];       //  9 KB W^T (f16)

    const int tid  = threadIdx.x;
    const int lane = tid & 31;
    const int w    = tid >> 5;
    const int b    = blockIdx.x >> 5;
    const int row0 = (blockIdx.x & 31) * 64;
    const int mg   = w & 3;          // M-group within block
    const int kh   = w >> 2;         // K-half (0 or 1)
    const int l16  = lane & 15;
    const int hi8  = (lane >> 4) << 3;   // K-offset 0 / 8 per WMMA 16-bit layout

    // ---- one-time staging: dinv[b][:], W^T as f16 ----
    for (int i = tid; i < NQ; i += 256) s_dinv[i] = dinv_g[b * NQ + i];
    for (int i = tid; i < 64 * 64; i += 256) {
        int o = i >> 6, f = i & 63;
        s_wt[o * WT_STRIDE + f] = (_Float16)W[f * 64 + o];
    }
    __syncthreads();

    const float* xb = x + (size_t)b * NQ * FQ;

    // Stage x chunk (32 k-rows for BOTH K-halves) into LDS, transposed f-major,
    // converted to f16 with dinv[k] folded in. 256 threads * 4 float4 each.
    auto stage = [&](int cIdx) {
        _Float16* buf0 = s_xs[0][cIdx & 1];
        _Float16* buf1 = s_xs[1][cIdx & 1];
        #pragma unroll
        for (int i = 0; i < 4; ++i) {
            int q    = i * 256 + tid;        // 0..1023
            int hf   = q >> 9;               // K-half
            int rem  = q & 511;
            int kk   = rem >> 4;             // k within chunk (0..31)
            int fq   = rem & 15;             // float4 index along f
            int k    = hf * 1024 + cIdx * 32 + kk;
            float4 v = *(const float4*)(xb + (size_t)k * FQ + fq * 4);
            float sc = s_dinv[k];
            _Float16* dst = (hf ? buf1 : buf0) + (fq * 4) * XS_STRIDE + kk;
            dst[0 * XS_STRIDE] = (_Float16)(v.x * sc);
            dst[1 * XS_STRIDE] = (_Float16)(v.y * sc);
            dst[2 * XS_STRIDE] = (_Float16)(v.z * sc);
            dst[3 * XS_STRIDE] = (_Float16)(v.w * sc);
        }
    };

    const int   rloc  = row0 + mg * 16 + l16;          // my A-matrix row (both half-waves)
    const float dinvr = s_dinv[rloc];
    const float* drow = dist + ((size_t)b * NQ + rloc) * NQ;

    v8f acc[4];
    #pragma unroll
    for (int t = 0; t < 4; ++t)
        #pragma unroll
        for (int j = 0; j < 8; ++j) acc[t][j] = 0.0f;

    stage(0);
    __syncthreads();

    // ---- main K loop: 32 chunks of K=32 per K-half ----
    for (int c = 0; c < 32; ++c) {
        if (c + 1 < 32) stage(c + 1);     // double-buffered prefetch (other buffer)

        const int k0 = kh * 1024 + c * 32;
        const int ka = k0 + hi8;
        // A fragment: row rloc, K = {ka..ka+7, ka+16..ka+23} (doc'd 16-bit A layout)
        float4 d0 = *(const float4*)(drow + ka);
        float4 d1 = *(const float4*)(drow + ka + 4);
        float4 d2 = *(const float4*)(drow + ka + 16);
        float4 d3 = *(const float4*)(drow + ka + 20);
        float dd[16];
        *(float4*)(dd + 0)  = d0; *(float4*)(dd + 4)  = d1;
        *(float4*)(dd + 8)  = d2; *(float4*)(dd + 12) = d3;
        v16h af;
        #pragma unroll
        for (int j = 0; j < 16; ++j)
            af[j] = (_Float16)(__expf(-0.2f * dd[j]) * dinvr);

        const _Float16* xsb = s_xs[kh][c & 1];
        #pragma unroll
        for (int t = 0; t < 4; ++t) {
            const _Float16* bp = xsb + (t * 16 + l16) * XS_STRIDE + hi8;
            v16h bf = cat8(*(const v8h*)bp, *(const v8h*)(bp + 16));
            acc[t] = __builtin_amdgcn_wmma_f32_16x16x32_f16(
                         false, af, false, bf, (short)0, acc[t], false, false);
        }
        __syncthreads();
    }

    // ---- reduce K-halves, then h @ W ----
    // Write partial h (C layout: VGPR v -> row v / v+8) to LDS row-major f32.
    float* hp = s_hp[w];
    #pragma unroll
    for (int t = 0; t < 4; ++t)
        #pragma unroll
        for (int vv = 0; vv < 8; ++vv)
            hp[(vv + hi8) * HP_STRIDE + t * 16 + l16] = acc[t][vv];
    __syncthreads();

    if (w < 4) {   // wave-uniform: EXEC stays all-ones inside (WMMA requirement)
        const float* p0 = s_hp[w];
        const float* p1 = s_hp[w + 4];
        v8f oacc[4];
        #pragma unroll
        for (int t = 0; t < 4; ++t)
            #pragma unroll
            for (int j = 0; j < 8; ++j) oacc[t][j] = 0.0f;

        #pragma unroll
        for (int kc2 = 0; kc2 < 64; kc2 += 32) {
            const int ka2 = kc2 + hi8;
            const float* r0p = p0 + l16 * HP_STRIDE + ka2;
            const float* r1p = p1 + l16 * HP_STRIDE + ka2;
            float ha[16], hb[16];
            *(float4*)(ha + 0)  = *(const float4*)(r0p);
            *(float4*)(ha + 4)  = *(const float4*)(r0p + 4);
            *(float4*)(ha + 8)  = *(const float4*)(r0p + 16);
            *(float4*)(ha + 12) = *(const float4*)(r0p + 20);
            *(float4*)(hb + 0)  = *(const float4*)(r1p);
            *(float4*)(hb + 4)  = *(const float4*)(r1p + 4);
            *(float4*)(hb + 8)  = *(const float4*)(r1p + 16);
            *(float4*)(hb + 12) = *(const float4*)(r1p + 20);
            v16h af2;
            #pragma unroll
            for (int j = 0; j < 16; ++j) af2[j] = (_Float16)(ha[j] + hb[j]);

            #pragma unroll
            for (int t2 = 0; t2 < 4; ++t2) {
                const _Float16* wp = s_wt + (t2 * 16 + l16) * WT_STRIDE + ka2;
                v16h bf = cat8(*(const v8h*)wp, *(const v8h*)(wp + 16));
                oacc[t2] = __builtin_amdgcn_wmma_f32_16x16x32_f16(
                               false, af2, false, bf, (short)0, oacc[t2], false, false);
            }
        }

        float* ob = out + ((size_t)b * NQ + row0 + w * 16) * FQ;
        #pragma unroll
        for (int t2 = 0; t2 < 4; ++t2)
            #pragma unroll
            for (int vv = 0; vv < 8; ++vv)
                ob[(vv + hi8) * FQ + t2 * 16 + l16] = oacc[t2][vv];
    }
}

// ---------------------------------------------------------------------------
extern "C" void kernel_launch(void* const* d_in, const int* in_sizes, int n_in,
                              void* d_out, int out_size, void* d_ws, size_t ws_size,
                              hipStream_t stream) {
    const float* x    = (const float*)d_in[0];   // [8,2048,64]
    const float* dist = (const float*)d_in[1];   // [8,2048,2048]
    const float* W    = (const float*)d_in[2];   // [64,64]
    float* out  = (float*)d_out;                 // [8,2048,64]
    float* dinv = (float*)d_ws;                  // 8*2048 f32 = 64 KB scratch

    // Phase 1: one wave per row, 8 rows per block -> 16384/8 = 2048 blocks
    dgc_degree_kernel<<<dim3(2048), dim3(256), 0, stream>>>(dist, dinv);
    // Phase 2: 8 batches * 32 row-tiles = 256 blocks of 256 threads
    dgc_gcn_kernel<<<dim3(256), dim3(256), 0, stream>>>(x, dist, W, dinv, out);
}
